// RoIFilter_73400991088925
// MI455X (gfx1250) — matrix operand order repaired and checked by reading, not compiled, not run on hardware
//
#include <hip/hip_runtime.h>
#include <hip/hip_bf16.h>
#include <math.h>

typedef _Float16 v16h __attribute__((ext_vector_type(16)));
typedef float    v8f  __attribute__((ext_vector_type(8)));

#define D_MODEL 512
#define S_SLOTS 64
#define B_DIM   4
#define L_DIM   8192
#define BL      (B_DIM * L_DIM)        // 32768 rows
#define NCHUNK  16
#define CHUNK_L (L_DIM / NCHUNK)       // 512

// workspace layout (in floats)
#define WS_LOGITS 0
#define WS_PMAX  (BL * S_SLOTS)                          // 2,097,152
#define WS_PSUM  (WS_PMAX + B_DIM * S_SLOTS * NCHUNK)    // +4096
#define WS_CMAX  (WS_PSUM + B_DIM * S_SLOTS * NCHUNK)    // +4096
#define WS_CSUM  (WS_CMAX + B_DIM * S_SLOTS)             // +256

// ---------------------------------------------------------------------------
// Kernel 1: logits[b,l,s] = q[b,l,:] . W_mk[s,:]   via v_wmma_f32_16x16x32_f16
// One wave -> 16 rows x 64 slots. W_mk staged in LDS as f16 (64 KB).
// ---------------------------------------------------------------------------
__global__ __launch_bounds__(256) void k1_gemm_qk(const float* __restrict__ q,
                                                  const float* __restrict__ wmk,
                                                  float* __restrict__ logits) {
    __shared__ __attribute__((aligned(32))) _Float16 lds_wmk[S_SLOTS * D_MODEL]; // 64 KB
    for (int i = threadIdx.x; i < S_SLOTS * D_MODEL; i += 256)
        lds_wmk[i] = (_Float16)wmk[i];
    __syncthreads();

    const int wave     = (blockIdx.x * 256 + threadIdx.x) >> 5;
    const int lane     = threadIdx.x & 31;
    const int rowStart = wave * 16;
    const int m        = lane & 15;            // A row / C col index
    const int kb       = (lane >> 4) << 3;     // C-frag row base: 0 or 8
    const int chunk    = (lane >> 4) << 4;     // B-frag K base: 0 or 16
    const float* qrow  = q + (size_t)(rowStart + m) * D_MODEL;

    v8f acc[4] = {v8f{}, v8f{}, v8f{}, v8f{}};

    for (int kk = 0; kk < D_MODEL; kk += 32) {
        v16h a;
#pragma unroll
        for (int i = 0; i < 8; ++i) a[i]     = (_Float16)qrow[kk + chunk + i];
#pragma unroll
        for (int i = 0; i < 8; ++i) a[8 + i] = (_Float16)qrow[kk + chunk + 8 + i];
#pragma unroll
        for (int nt = 0; nt < 4; ++nt) {
            const int s = nt * 16 + m;
            const v16h bm = *(const v16h*)&lds_wmk[s * D_MODEL + kk + chunk];
            acc[nt] = __builtin_amdgcn_wmma_f32_16x16x32_f16(
                false, a, false, bm, (short)0, acc[nt], false, false);
        }
    }

#pragma unroll
    for (int nt = 0; nt < 4; ++nt) {
        const int s = nt * 16 + m;
#pragma unroll
        for (int j = 0; j < 8; ++j) {
            const int row = rowStart + kb + j;   // M = j + 8*(lane>=16)
            logits[(size_t)row * S_SLOTS + s] = acc[nt][j];
        }
    }
}

// ---------------------------------------------------------------------------
// Kernel 2a: per-(b, L-chunk) partial max & sum(exp) for softmax over L.
// Threads cover all 64 slots contiguously -> fully coalesced logits reads.
// ---------------------------------------------------------------------------
__global__ __launch_bounds__(256) void k2a_partials(const float* __restrict__ logits,
                                                    float* __restrict__ pMax,
                                                    float* __restrict__ pSum) {
    __shared__ float red[4][S_SLOTS];
    const int b     = blockIdx.x >> 4;
    const int chunk = blockIdx.x & 15;
    const int s     = threadIdx.x & 63;
    const int lg    = threadIdx.x >> 6;
    const size_t base = ((size_t)b * L_DIM + (size_t)chunk * CHUNK_L) * S_SLOTS + s;

    float mx = -3.0e38f;
    for (int it = 0; it < CHUNK_L / 4; ++it)
        mx = fmaxf(mx, logits[base + (size_t)(it * 4 + lg) * S_SLOTS]);
    red[lg][s] = mx;
    __syncthreads();
    const float cm = fmaxf(fmaxf(red[0][s], red[1][s]), fmaxf(red[2][s], red[3][s]));
    __syncthreads();

    float sum = 0.f;
    for (int it = 0; it < CHUNK_L / 4; ++it)
        sum += expf(logits[base + (size_t)(it * 4 + lg) * S_SLOTS] - cm);
    red[lg][s] = sum;
    __syncthreads();
    if (threadIdx.x < 64) {
        const float tot = red[0][s] + red[1][s] + red[2][s] + red[3][s];
        const int bs = b * S_SLOTS + s;
        pMax[bs * NCHUNK + chunk] = cm;
        pSum[bs * NCHUNK + chunk] = tot;
    }
}

// Kernel 2b: flash-style merge of the 16 partials per (b,s) column.
__global__ __launch_bounds__(256) void k2b_combine(const float* __restrict__ pMax,
                                                   const float* __restrict__ pSum,
                                                   float* __restrict__ cMax,
                                                   float* __restrict__ cSum) {
    const int bs = threadIdx.x;     // 256 columns, single block
    float gm = -3.0e38f;
    for (int c = 0; c < NCHUNK; ++c) gm = fmaxf(gm, pMax[bs * NCHUNK + c]);
    float gs = 0.f;
    for (int c = 0; c < NCHUNK; ++c) gs += pSum[bs * NCHUNK + c] * expf(pMax[bs * NCHUNK + c] - gm);
    cMax[bs] = gm;
    cSum[bs] = gs;
}

// ---------------------------------------------------------------------------
// Kernel 3: attn = softmax_L / rowsum_S, out = attn . W_mv^T (WMMA),
// sigmoid/mean/threshold, masked copy of queries. One wave -> 16 rows.
// ---------------------------------------------------------------------------
__global__ __launch_bounds__(256) void k3_attend(const float* __restrict__ q,
                                                 const float* __restrict__ wmv,
                                                 const float* __restrict__ logits,
                                                 const float* __restrict__ cMax,
                                                 const float* __restrict__ cSum,
                                                 const float* __restrict__ thrPtr,
                                                 float* __restrict__ out) {
    __shared__ __attribute__((aligned(32))) _Float16 lds_wmv[D_MODEL * S_SLOTS]; // 64 KB
    for (int i = threadIdx.x; i < D_MODEL * S_SLOTS; i += 256)
        lds_wmv[i] = (_Float16)wmv[i];
    __syncthreads();

    const int wave     = (blockIdx.x * 256 + threadIdx.x) >> 5;
    const int lane     = threadIdx.x & 31;
    const int rowStart = wave * 16;
    const int m        = lane & 15;
    const int kb       = (lane >> 4) << 3;    // 0 or 8
    const int chunk    = (lane >> 4) << 4;    // 0 or 16
    const int r        = rowStart + m;
    const int b        = r >> 13;             // r / 8192
    const float* lrow  = logits + (size_t)r * S_SLOTS;
    const float* cm    = cMax + b * S_SLOTS;
    const float* cs    = cSum + b * S_SLOTS;

    // exp weights for this lane's 32 of the 64 slots (exactly its A-frag set)
    float ew[32];
    float part = 0.f;
#pragma unroll
    for (int c = 0; c < 4; ++c) {
#pragma unroll
        for (int u = 0; u < 8; ++u) {
            const int s = chunk + (c >> 1) * 32 + (c & 1) * 8 + u;
            const float e = expf(lrow[s] - cm[s]) / cs[s];
            ew[c * 8 + u] = e;
            part += e;
        }
    }
    const float rs  = part + __shfl_xor(part, 16, 32);  // complementary half
    const float inv = 1.0f / rs;

    v16h a0, a1;
#pragma unroll
    for (int i = 0; i < 16; ++i) a0[i] = (_Float16)(ew[i] * inv);
#pragma unroll
    for (int i = 0; i < 16; ++i) a1[i] = (_Float16)(ew[16 + i] * inv);

    float rowPart[8];
#pragma unroll
    for (int j = 0; j < 8; ++j) rowPart[j] = 0.f;

    for (int nt = 0; nt < 32; ++nt) {
        const int d = nt * 16 + m;
        const v16h b0 = *(const v16h*)&lds_wmv[d * S_SLOTS + chunk];
        const v16h b1 = *(const v16h*)&lds_wmv[d * S_SLOTS + 32 + chunk];
        v8f acc = {};
        acc = __builtin_amdgcn_wmma_f32_16x16x32_f16(false, a0, false, b0, (short)0, acc, false, false);
        acc = __builtin_amdgcn_wmma_f32_16x16x32_f16(false, a1, false, b1, (short)0, acc, false, false);
#pragma unroll
        for (int j = 0; j < 8; ++j)
            rowPart[j] += 1.0f / (1.0f + expf(-acc[j]));
    }

    // reduce sigmoid sums across the 16 lanes sharing each row set
#pragma unroll
    for (int off = 1; off < 16; off <<= 1)
#pragma unroll
        for (int j = 0; j < 8; ++j)
            rowPart[j] += __shfl_xor(rowPart[j], off, 16);

    const float thr = thrPtr[0];
    int bits = 0;
#pragma unroll
    for (int j = 0; j < 8; ++j)
        if (rowPart[j] * (1.0f / (float)D_MODEL) >= thr) bits |= (1 << j);
    const int lo = __shfl(bits, 0, 32);    // rows 0..7
    const int hi = __shfl(bits, 16, 32);   // rows 8..15
    const int maskWord = (lo & 0xFF) | ((hi & 0xFF) << 8);

    // masked streaming copy of the 16 rows (float4, coalesced)
    const float4* q4 = (const float4*)(q + (size_t)rowStart * D_MODEL);
    float4*       o4 = (float4*)(out + (size_t)rowStart * D_MODEL);
    for (int i = 0; i < 64; ++i) {
        const int vi = i * 32 + lane;      // 0..2047 (128 float4 per row)
        const int mm = vi >> 7;            // row within tile
        const float sc = ((maskWord >> mm) & 1) ? 1.0f : 0.0f;
        const float4 v = q4[vi];
        float4 w;
        w.x = v.x * sc; w.y = v.y * sc; w.z = v.z * sc; w.w = v.w * sc;
        o4[vi] = w;
    }
}

extern "C" void kernel_launch(void* const* d_in, const int* in_sizes, int n_in,
                              void* d_out, int out_size, void* d_ws, size_t ws_size,
                              hipStream_t stream) {
    const float* q   = (const float*)d_in[0];
    const float* wmk = (const float*)d_in[1];
    const float* wmv = (const float*)d_in[2];
    const float* thr = (const float*)d_in[3];
    float* out = (float*)d_out;
    float* ws  = (float*)d_ws;

    float* logits = ws + WS_LOGITS;
    float* pMax   = ws + WS_PMAX;
    float* pSum   = ws + WS_PSUM;
    float* cMax   = ws + WS_CMAX;
    float* cSum   = ws + WS_CSUM;

    k1_gemm_qk <<<BL / 128, 256, 0, stream>>>(q, wmk, logits);          // 256 blocks, 8 waves each
    k2a_partials<<<B_DIM * NCHUNK, 256, 0, stream>>>(logits, pMax, pSum);
    k2b_combine <<<1, 256, 0, stream>>>(pMax, pSum, cMax, cSum);
    k3_attend   <<<BL / 128, 256, 0, stream>>>(q, wmv, logits, cMax, cSum, thr, out);
}
